// CausalMolSSM_47244640256225
// MI455X (gfx1250) — matrix operand order repaired and checked by reading, not compiled, not run on hardware
//
#include <hip/hip_runtime.h>

#define LSEQ 1024
#define DM   512
#define DI   1024
#define DS   16
#define PN   (DI + 4*DS)   // 1088

typedef unsigned short ushort_t;
typedef __attribute__((ext_vector_type(16))) __bf16 v16bf;
typedef __attribute__((ext_vector_type(8)))  float  v8f;

union Frag { uint4 u[2]; v16bf v; };

__device__ __forceinline__ ushort_t f2bf(float f) {
    unsigned int u = __float_as_uint(f);
    u += 0x7FFFu + ((u >> 16) & 1u);   // round-to-nearest-even
    return (ushort_t)(u >> 16);
}

// ---------------- conversion kernels ----------------
__global__ void k_f32_to_bf16(const float* __restrict__ s, ushort_t* __restrict__ d, int n) {
    int i = blockIdx.x * blockDim.x + threadIdx.x;
    if (i < n) d[i] = f2bf(s[i]);
}

__global__ void k_flip_to_bf16(const float* __restrict__ s, ushort_t* __restrict__ d,
                               int rows, int cols) {
    int i = blockIdx.x * blockDim.x + threadIdx.x;
    if (i < rows * cols) {
        int r = i / cols, c = i - r * cols;
        d[i] = f2bf(s[(size_t)(rows - 1 - r) * cols + c]);
    }
}

// ---------------- WMMA bf16 GEMM: C[M,N] = A[M,K] * W[N,K]^T (+bias) ----------------
// one wave -> one 32x32 tile of C (2x2 fragment blocking, 4 WMMAs per K-step);
// 2 waves/block along N.
__global__ void k_gemm_bf16(const ushort_t* __restrict__ A, int lda,
                            const ushort_t* __restrict__ W, int ldw,
                            const float* __restrict__ bias,
                            float* __restrict__ Cf, ushort_t* __restrict__ Cb, int ldc,
                            int M, int N, int K) {
    const int wave = threadIdx.x >> 5;
    const int lane = threadIdx.x & 31;
    const int nt = blockIdx.x * 2 + wave;   // 32-col tile index
    const int mt = blockIdx.y;              // 32-row tile index
    if (nt * 32 >= N || mt * 32 >= M) return;

    const int row = lane & 15;
    const int hi  = lane >> 4;

    const ushort_t* Ap0 = A + (size_t)(mt * 32 + row) * lda;
    const ushort_t* Ap1 = Ap0 + (size_t)16 * lda;
    const ushort_t* Wp0 = W + (size_t)(nt * 32 + row) * ldw;
    const ushort_t* Wp1 = Wp0 + (size_t)16 * ldw;

    v8f c00 = {}, c01 = {}, c10 = {}, c11 = {};
    for (int k = 0; k < K; k += 32) {
        __builtin_prefetch(Ap0 + k + 256, 0, 1);
        __builtin_prefetch(Wp0 + k + 256, 0, 1);
        Frag a0, a1, b0, b1;
        // A 16x32 bf16 fragment: lanes 0-15 hold K={0..7,16..23}, lanes 16-31 K={8..15,24..31}
        a0.u[0] = *(const uint4*)(Ap0 + k + hi * 8);
        a0.u[1] = *(const uint4*)(Ap0 + k + 16 + hi * 8);
        a1.u[0] = *(const uint4*)(Ap1 + k + hi * 8);
        a1.u[1] = *(const uint4*)(Ap1 + k + 16 + hi * 8);
        // B 32x16 bf16 fragment (from row-major W[N,K]): lanes 0-15 K={0..7,16..23}, lanes 16-31 K={8..15,24..31}
        b0.u[0] = *(const uint4*)(Wp0 + k + hi * 8);
        b0.u[1] = *(const uint4*)(Wp0 + k + 16 + hi * 8);
        b1.u[0] = *(const uint4*)(Wp1 + k + hi * 8);
        b1.u[1] = *(const uint4*)(Wp1 + k + 16 + hi * 8);
        c00 = __builtin_amdgcn_wmma_f32_16x16x32_bf16(false, a0.v, false, b0.v,
                                                      (short)0, c00, false, false);
        c01 = __builtin_amdgcn_wmma_f32_16x16x32_bf16(false, a0.v, false, b1.v,
                                                      (short)0, c01, false, false);
        c10 = __builtin_amdgcn_wmma_f32_16x16x32_bf16(false, a1.v, false, b0.v,
                                                      (short)0, c10, false, false);
        c11 = __builtin_amdgcn_wmma_f32_16x16x32_bf16(false, a1.v, false, b1.v,
                                                      (short)0, c11, false, false);
    }

    const int nn0 = nt * 32 + row;
    const int nn1 = nn0 + 16;
    const float bv0 = bias ? bias[nn0] : 0.f;
    const float bv1 = bias ? bias[nn1] : 0.f;
#pragma unroll
    for (int r = 0; r < 8; ++r) {
        // C layout: VGPR r -> M=r for lanes 0-15, M=r+8 for lanes 16-31; N=lane&15
        const int mm0 = mt * 32 + r + hi * 8;
        const int mm1 = mm0 + 16;
        const float v00 = c00[r] + bv0;
        const float v01 = c01[r] + bv1;
        const float v10 = c10[r] + bv0;
        const float v11 = c11[r] + bv1;
        if (Cf) {
            Cf[(size_t)mm0 * ldc + nn0] = v00;
            Cf[(size_t)mm0 * ldc + nn1] = v01;
            Cf[(size_t)mm1 * ldc + nn0] = v10;
            Cf[(size_t)mm1 * ldc + nn1] = v11;
        }
        if (Cb) {
            Cb[(size_t)mm0 * ldc + nn0] = f2bf(v00);
            Cb[(size_t)mm0 * ldc + nn1] = f2bf(v01);
            Cb[(size_t)mm1 * ldc + nn0] = f2bf(v10);
            Cb[(size_t)mm1 * ldc + nn1] = f2bf(v11);
        }
    }
}

// ---------------- causal depthwise conv (width 4) + SiLU ----------------
__global__ void k_conv_silu(const float* __restrict__ xz, const float* __restrict__ cw,
                            const float* __restrict__ cb,
                            float* __restrict__ xs_f, ushort_t* __restrict__ xs_b) {
    int i = blockIdx.x * blockDim.x + threadIdx.x;
    if (i >= LSEQ * DI) return;
    int l = i >> 10, d = i & (DI - 1);
    float acc = cb[d];
#pragma unroll
    for (int j = 0; j < 4; ++j) {
        int ls = l - 3 + j;
        if (ls >= 0) acc += cw[d * 4 + j] * xz[(size_t)ls * (2 * DI) + d];
    }
    float s = acc / (1.f + __expf(-acc));
    xs_f[i] = s;
    xs_b[i] = f2bf(s);
}

// ---------------- sequential complex SSM scan (bilinear form) + gate ----------------
// thread t -> channel d = t>>4, state n = t&15; reduce over n with intra-16-lane shuffles
__global__ void k_scan(const float* __restrict__ dtp, const float* __restrict__ p,
                       const float* __restrict__ xs, const float* __restrict__ xz,
                       const float* __restrict__ alr, const float* __restrict__ ali,
                       const float* __restrict__ Dv, ushort_t* __restrict__ yg) {
    int t = blockIdx.x * blockDim.x + threadIdx.x;
    if (t >= DI * DS) return;
    int d = t >> 4, n = t & 15;

    float ea  = __expf(alr[d * DS + n]);
    float Are = -ea * __cosf(ali[d * DS + n]);
    float Aim = -ea * __sinf(ali[d * DS + n]);
    float Dd  = Dv[d];

    float hre = 0.f, him = 0.f;
    for (int l = 0; l < LSEQ; ++l) {
        float dp    = dtp[(size_t)l * DI + d];
        float delta = dp > 20.f ? dp : log1pf(__expf(dp));   // softplus
        float dre = delta * Are, dim = delta * Aim;
        float den_re = 2.f - dre, den_im = -dim;
        float inv = 1.f / (den_re * den_re + den_im * den_im + 1e-18f);
        float nre = 2.f + dre, nim = dim;
        float Abr = (nre * den_re + nim * den_im) * inv;     // A_bar = numer/denom
        float Abi = (nim * den_re - nre * den_im) * inv;
        float g   = 2.f * delta * inv;                        // 2*dt/denom
        float gre = g * den_re, gim = -g * den_im;

        const float* pr = p + (size_t)l * PN;
        float Bre = pr[DI + n],          Bim = pr[DI + DS + n];
        float Cre = pr[DI + 2 * DS + n], Cim = pr[DI + 3 * DS + n];
        float u = xs[(size_t)l * DI + d];

        float ubre = (gre * Bre - gim * Bim) * u;
        float ubim = (gre * Bim + gim * Bre) * u;
        float hr2 = Abr * hre - Abi * him + ubre;
        float hi2 = Abr * him + Abi * hre + ubim;
        hre = hr2; him = hi2;

        float contrib = Cre * hre - Cim * him;                // Re(C*h)
        contrib += __shfl_xor(contrib, 8, 16);
        contrib += __shfl_xor(contrib, 4, 16);
        contrib += __shfl_xor(contrib, 2, 16);
        contrib += __shfl_xor(contrib, 1, 16);
        if (n == 0) {
            float y = contrib + Dd * u;
            float z = xz[(size_t)l * (2 * DI) + DI + d];
            float gate = z / (1.f + __expf(-z));
            yg[(size_t)l * DI + d] = f2bf(y * gate);
        }
    }
}

// ---------------- concat fwd + flipped bwd -> bf16 ----------------
__global__ void k_cat(const float* __restrict__ f, const float* __restrict__ b,
                      ushort_t* __restrict__ cat) {
    int i = blockIdx.x * blockDim.x + threadIdx.x;
    if (i >= LSEQ * 2 * DM) return;
    int l = i >> 10, c = i & 1023;
    float v = (c < DM) ? f[(size_t)l * DM + c]
                       : b[(size_t)(LSEQ - 1 - l) * DM + (c - DM)];
    cat[i] = f2bf(v);
}

// ---------------- host launch ----------------
extern "C" void kernel_launch(void* const* d_in, const int* in_sizes, int n_in,
                              void* d_out, int out_size, void* d_ws, size_t ws_size,
                              hipStream_t stream) {
    const float* x = (const float*)d_in[0];
    const float* in_w[2]   = {(const float*)d_in[1],  (const float*)d_in[11]};
    const float* conv_w[2] = {(const float*)d_in[2],  (const float*)d_in[12]};
    const float* conv_b[2] = {(const float*)d_in[3],  (const float*)d_in[13]};
    const float* xp_w[2]   = {(const float*)d_in[4],  (const float*)d_in[14]};
    const float* dt_w[2]   = {(const float*)d_in[5],  (const float*)d_in[15]};
    const float* dt_b[2]   = {(const float*)d_in[6],  (const float*)d_in[16]};
    const float* alr = (const float*)d_in[7];
    const float* ali = (const float*)d_in[8];
    const float* Dv  = (const float*)d_in[9];
    const float* out_w[2]  = {(const float*)d_in[10], (const float*)d_in[17]};
    const float* fu_w = (const float*)d_in[18];
    const float* fu_b = (const float*)d_in[19];

    char* base = (char*)d_ws;
    size_t off = 0;
    auto alloc = [&](size_t bytes) -> void* {
        void* p = base + off;
        off = (off + bytes + 255) & ~(size_t)255;
        return p;
    };

    ushort_t* x_bf     = (ushort_t*)alloc((size_t)LSEQ * DM * 2);
    ushort_t* xflip_bf = (ushort_t*)alloc((size_t)LSEQ * DM * 2);
    ushort_t* w_in_bf[2], *w_xp_bf[2], *w_dt_bf[2], *w_out_bf[2];
    float *xz_f[2], *xssm_f[2], *p_f[2], *dtp_f[2], *brout_f[2];
    ushort_t *xssm_bf[2], *p_bf[2], *yg_bf[2];
    for (int b = 0; b < 2; ++b) {
        w_in_bf[b]  = (ushort_t*)alloc((size_t)2 * DI * DM * 2);
        w_xp_bf[b]  = (ushort_t*)alloc((size_t)PN * DI * 2);
        w_dt_bf[b]  = (ushort_t*)alloc((size_t)DI * DI * 2);
        w_out_bf[b] = (ushort_t*)alloc((size_t)DM * DI * 2);
        xz_f[b]     = (float*)   alloc((size_t)LSEQ * 2 * DI * 4);
        xssm_f[b]   = (float*)   alloc((size_t)LSEQ * DI * 4);
        xssm_bf[b]  = (ushort_t*)alloc((size_t)LSEQ * DI * 2);
        p_f[b]      = (float*)   alloc((size_t)LSEQ * PN * 4);
        p_bf[b]     = (ushort_t*)alloc((size_t)LSEQ * PN * 2);
        dtp_f[b]    = (float*)   alloc((size_t)LSEQ * DI * 4);
        yg_bf[b]    = (ushort_t*)alloc((size_t)LSEQ * DI * 2);
        brout_f[b]  = (float*)   alloc((size_t)LSEQ * DM * 4);
    }
    ushort_t* w_fu_bf = (ushort_t*)alloc((size_t)DM * 2 * DM * 2);
    ushort_t* cat_bf  = (ushort_t*)alloc((size_t)LSEQ * 2 * DM * 2);

    auto cvt = [&](const float* s, ushort_t* d, int n) {
        k_f32_to_bf16<<<(n + 255) / 256, 256, 0, stream>>>(s, d, n);
    };
    auto gemm = [&](const ushort_t* A, int lda, const ushort_t* W, int ldw,
                    const float* bias, float* Cf, ushort_t* Cb, int ldc,
                    int M, int N, int K) {
        dim3 g(N / 64, M / 32);           // 2 waves/block along N, 32x32 tile per wave
        k_gemm_bf16<<<g, 64, 0, stream>>>(A, lda, W, ldw, bias, Cf, Cb, ldc, M, N, K);
    };

    // weight + input conversions
    cvt(x, x_bf, LSEQ * DM);
    k_flip_to_bf16<<<(LSEQ * DM + 255) / 256, 256, 0, stream>>>(x, xflip_bf, LSEQ, DM);
    for (int b = 0; b < 2; ++b) {
        cvt(in_w[b],  w_in_bf[b],  2 * DI * DM);
        cvt(xp_w[b],  w_xp_bf[b],  PN * DI);
        cvt(dt_w[b],  w_dt_bf[b],  DI * DI);
        cvt(out_w[b], w_out_bf[b], DM * DI);
    }
    cvt(fu_w, w_fu_bf, DM * 2 * DM);

    for (int b = 0; b < 2; ++b) {
        const ushort_t* xin = b ? xflip_bf : x_bf;
        // xz = x @ in_w^T   [1024 x 2048]
        gemm(xin, DM, w_in_bf[b], DM, nullptr, xz_f[b], nullptr, 2 * DI,
             LSEQ, 2 * DI, DM);
        // depthwise conv + SiLU
        k_conv_silu<<<(LSEQ * DI + 255) / 256, 256, 0, stream>>>(
            xz_f[b], conv_w[b], conv_b[b], xssm_f[b], xssm_bf[b]);
        // p = x_ssm @ xp_w^T   [1024 x 1088]
        gemm(xssm_bf[b], DI, w_xp_bf[b], DI, nullptr, p_f[b], p_bf[b], PN,
             LSEQ, PN, DI);
        // delta_pre = p[:, :DI] @ dt_w^T + dt_b   [1024 x 1024]
        gemm(p_bf[b], PN, w_dt_bf[b], DI, dt_b[b], dtp_f[b], nullptr, DI,
             LSEQ, DI, DI);
        // complex SSM scan + D skip + SiLU(z) gate
        k_scan<<<(DI * DS + 255) / 256, 256, 0, stream>>>(
            dtp_f[b], p_f[b], xssm_f[b], xz_f[b], alr, ali, Dv, yg_bf[b]);
        // branch output = yg @ out_w^T   [1024 x 512]
        gemm(yg_bf[b], DI, w_out_bf[b], DI, nullptr, brout_f[b], nullptr, DM,
             LSEQ, DM, DI);
    }

    // concat [fwd, flip(bwd)] -> bf16
    k_cat<<<(LSEQ * 2 * DM + 255) / 256, 256, 0, stream>>>(brout_f[0], brout_f[1], cat_bf);
    // final fusion GEMM + bias -> d_out   [1024 x 512] f32
    gemm(cat_bf, 2 * DM, w_fu_bf, 2 * DM, fu_b, (float*)d_out, nullptr, DM,
         LSEQ, DM, 2 * DM);
}
